// MultiHeadAttention_4329327034628
// MI455X (gfx1250) — compile-verified, matrix operand order checked by reading, not run on hardware
//
#include <hip/hip_runtime.h>

// ---------------------------------------------------------------- types
typedef __bf16 bf16_t;
typedef __attribute__((ext_vector_type(16))) __bf16 v16bf;
typedef __attribute__((ext_vector_type(8)))  __bf16 v8bf;
typedef __attribute__((ext_vector_type(8)))  float  v8f;
typedef __attribute__((ext_vector_type(4)))  unsigned int v4u;
typedef __attribute__((ext_vector_type(8)))  int v8i;
typedef __attribute__((ext_vector_type(4)))  int v4i;

#define BATCH 2
#define QLEN  2048
#define HD    768          // H * DKV = model dim
#define NH    12
#define DK    64
#define MTOT  (BATCH * QLEN)   // 4096
#define NEGINF (-1.0e9f)

// ---------------------------------------------------------------- helpers
__device__ __forceinline__ v16bf load_frag(const bf16_t* __restrict__ p) {
  v8bf lo = *(const v8bf*)(p);
  v8bf hi = *(const v8bf*)(p + 16);
  v16bf r;
#pragma unroll
  for (int i = 0; i < 8; ++i) { r[i] = lo[i]; r[i + 8] = hi[i]; }
  return r;
}

__device__ __forceinline__ v8f wmma_bf16(v16bf a, v16bf b, v8f c) {
  return __builtin_amdgcn_wmma_f32_16x16x32_bf16(
      false, a, false, b, (short)0, c, false, false);
}

// DPP16 row-rotate (16-lane row == the half-wave owning one C-layout row
// group).  Pure-VALU all-reduce: no LDS traffic, no DScnt waits.
template <int CTRL>
__device__ __forceinline__ float dpp_rot(float v) {
  return __int_as_float(__builtin_amdgcn_update_dpp(
      0, __float_as_int(v), CTRL, 0xF, 0xF, true));
}
__device__ __forceinline__ float row_allmax(float v) {
  v = fmaxf(v, dpp_rot<0x128>(v));   // row_ror:8
  v = fmaxf(v, dpp_rot<0x124>(v));   // row_ror:4
  v = fmaxf(v, dpp_rot<0x122>(v));   // row_ror:2
  v = fmaxf(v, dpp_rot<0x121>(v));   // row_ror:1
  return v;
}
__device__ __forceinline__ float row_allsum(float v) {
  v += dpp_rot<0x128>(v);
  v += dpp_rot<0x124>(v);
  v += dpp_rot<0x122>(v);
  v += dpp_rot<0x121>(v);
  return v;
}

// ---------------------------------------------------------------- TDM
// Issue a TDM 2D tile load: 16 rows x 32 cols of f32 from a 2048-stride
// tensor into LDS.  D# bitfields per CDNA5 ISA ch.8 (group0/group1).
#if defined(__has_builtin)
#if __has_builtin(__builtin_amdgcn_tensor_load_to_lds)
#define HAVE_TDM 1
#endif
#endif
#ifndef HAVE_TDM
#define HAVE_TDM 0
#endif

__device__ __forceinline__ void tdm_load_bias_tile(const float* gptr,
                                                   unsigned lds_off) {
#if HAVE_TDM
  unsigned lo = (unsigned)(unsigned long long)gptr;
  unsigned hi = (unsigned)(((unsigned long long)gptr) >> 32);
  lo = (unsigned)__builtin_amdgcn_readfirstlane((int)lo);
  hi = (unsigned)__builtin_amdgcn_readfirstlane((int)hi);
  lds_off = (unsigned)__builtin_amdgcn_readfirstlane((int)lds_off);

  v4u g0;
  g0[0] = 1u;                                     // count=1, user desc
  g0[1] = lds_off;                                // lds_addr (bytes)
  g0[2] = lo;                                     // global_addr[31:0]
  g0[3] = (hi & 0x01FFFFFFu) | 0x80000000u;       // addr[56:32] | type=2

  v8i g1;
  g1[0] = 0x00020000;            // workgroup_mask=0, data_size=2 (4 bytes)
  g1[1] = (int)(2048u << 16);    // tensor_dim0 = 2048 (low 16 @ bit48)
  g1[2] = (int)(2048u << 16);    // tensor_dim1 = 2048 (low 16 @ bit80)
  g1[3] = (int)(32u << 16);      // tile_dim0  = 32   (@ bit112)
  g1[4] = 16;                    // tile_dim1  = 16, tile_dim2 = 0
  g1[5] = 2048;                  // tensor_dim0_stride = 2048
  g1[6] = 0;                     // stride hi / tensor_dim1_stride = 0
  g1[7] = 0;
  v4i z = {0, 0, 0, 0};
#if __clang_major__ >= 23
  v8i z8 = {0, 0, 0, 0, 0, 0, 0, 0};
  __builtin_amdgcn_tensor_load_to_lds(g0, g1, z, z, z8, 0);
#else
  __builtin_amdgcn_tensor_load_to_lds(g0, g1, z, z, 0);
#endif
#else
  (void)gptr; (void)lds_off;
#endif
}

// ---------------------------------------------------------------- prep
__global__ void cvt_f32_to_bf16(const float* __restrict__ x,
                                bf16_t* __restrict__ y, int n) {
  int i = blockIdx.x * blockDim.x + threadIdx.x;
  if (i < n) y[i] = (bf16_t)x[i];
}

// wt[n][k] = (bf16) w[k][n]   (768 x 768)
__global__ void transpose_w_bf16(const float* __restrict__ w,
                                 bf16_t* __restrict__ wt) {
  int i = blockIdx.x * blockDim.x + threadIdx.x;  // < 768*768
  int k = i / HD, n = i % HD;
  wt[(size_t)n * HD + k] = (bf16_t)w[i];
}

// ---------------------------------------------------------------- GEMM
// C[4096,768] = A[4096,768](bf16 rm) * Bt[768,768](bf16 [n][k])
// 128-thread block = 4 waves; wave tile 32x64 (2x4 WMMA accumulators)
// OUT_MODE 0: bf16 rm   1: bf16 per-head transposed (V)   2: f32 rm
template <int OUT_MODE>
__global__ __launch_bounds__(128) void gemm_bf16(
    const bf16_t* __restrict__ A, const bf16_t* __restrict__ Bt,
    void* __restrict__ Cout) {
  const int lane  = threadIdx.x & 31;
  const int wid   = threadIdx.x >> 5;
  const int waveM = wid & 1;          // 2 waves along M
  const int waveN = wid >> 1;         // 2 waves along N
  const int l15   = lane & 15;
  const int hi16  = lane >> 4;
  const int kb    = hi16 << 3;        // 0 or 8

  const int m_base = blockIdx.y * 64  + waveM * 32;
  const int n_base = blockIdx.x * 128 + waveN * 64;

  const bf16_t* a0p = A  + (size_t)(m_base +      l15) * HD + kb;
  const bf16_t* a1p = a0p + (size_t)16 * HD;
  const bf16_t* b0p = Bt + (size_t)(n_base +      l15) * HD + kb;

  v8f acc[2][4] = {};
  for (int kt = 0; kt < HD; kt += 32) {
    if (kt + 32 < HD) {                      // global_prefetch_b8
      __builtin_prefetch(a0p + kt + 32, 0, 3);
      __builtin_prefetch(b0p + kt + 32, 0, 3);
    }
    v16bf a0 = load_frag(a0p + kt);
    v16bf a1 = load_frag(a1p + kt);
#pragma unroll
    for (int tn = 0; tn < 4; ++tn) {
      v16bf b = load_frag(b0p + (size_t)(tn * 16) * HD + kt);
      acc[0][tn] = wmma_bf16(a0, b, acc[0][tn]);
      acc[1][tn] = wmma_bf16(a1, b, acc[1][tn]);
    }
  }

#pragma unroll
  for (int tm = 0; tm < 2; ++tm)
#pragma unroll
    for (int tn = 0; tn < 4; ++tn)
#pragma unroll
      for (int r = 0; r < 8; ++r) {
        int row = m_base + tm * 16 + r + 8 * hi16;
        int col = n_base + tn * 16 + l15;
        float v = acc[tm][tn][r];
        if (OUT_MODE == 0) {
          ((bf16_t*)Cout)[(size_t)row * HD + col] = (bf16_t)v;
        } else if (OUT_MODE == 1) {
          int b = row >> 11, q = row & 2047;     // row = b*2048 + q
          int h = col >> 6,  d = col & 63;       // col = h*64 + d
          ((bf16_t*)Cout)[((size_t)((b * NH + h) * DK + d) << 11) + q] =
              (bf16_t)v;
        } else {
          ((float*)Cout)[(size_t)row * HD + col] = v;
        }
      }
}

// ---------------------------------------------------------------- attention
// grid: (QLEN/64, BATCH*NH); block 128 = 4 waves; wave owns 16 q rows.
// position_bias tiles (the 192MB stream) are fetched by the Tensor Data
// Mover into per-wave double-buffered LDS, one k-tile ahead of compute.
__global__ __launch_bounds__(128) void attn_kernel(
    const bf16_t* __restrict__ qbf,   // [B*Q][768] bf16
    const bf16_t* __restrict__ kbf,   // [B*Q][768] bf16
    const bf16_t* __restrict__ vt,    // [(b*H+h)*64 + d][2048] bf16
    const float*  __restrict__ bias,  // [1][H][Q][Q] f32
    const float*  __restrict__ mask,  // [B][1][1][Q] f32
    bf16_t* __restrict__ obf) {       // [B*Q][768] bf16
  const int bh   = blockIdx.y;
  const int b    = bh / NH;
  const int h    = bh % NH;
  const int wave = threadIdx.x >> 5;
  const int lane = threadIdx.x & 31;
  const int l15  = lane & 15;
  const int hi16 = lane >> 4;
  const int kb   = hi16 << 3;

  const int qRow0 = blockIdx.x * 64 + wave * 16;   // local q of this wave

  __shared__ __align__(16) bf16_t lds_p[4][16][40];      // P transpose buffer
  __shared__ __align__(16) float  bias_t[4][2][16 * 32]; // TDM double buffer

  // Q A-fragments (persist over the k loop)
  const bf16_t* qp =
      qbf + ((size_t)(b * QLEN) + qRow0 + l15) * HD + h * DK + kb;
  v16bf aq0 = load_frag(qp);
  v16bf aq1 = load_frag(qp + 32);

  float m_run[8], l_run[8];
#pragma unroll
  for (int r = 0; r < 8; ++r) { m_run[r] = -1.0e30f; l_run[r] = 0.0f; }
  v8f o_acc[4] = {};

  const float* bias_base = bias + ((size_t)h * QLEN + qRow0) * QLEN;
  const int kend = qRow0 + 16;            // causal limit for this wave

  int buf = 0;
  tdm_load_bias_tile(bias_base /*kt=0*/,
                     (unsigned)(size_t)(void*)&bias_t[wave][0][0]);

  for (int kt = 0; kt < kend; kt += 32) {
    const bool more = (kt + 32) < kend;
    if (more)
      tdm_load_bias_tile(bias_base + kt + 32,
                         (unsigned)(size_t)(void*)&bias_t[wave][buf ^ 1][0]);

    // ---- S = Q K^T  (16 x 32 tile) -- overlaps with the TDM fetch
    const bf16_t* kp0 =
        kbf + ((size_t)(b * QLEN) + kt + l15) * HD + h * DK + kb;
    const bf16_t* kp1 = kp0 + (size_t)16 * HD;
    v8f s0 = {}, s1 = {};
    s0 = wmma_bf16(aq0, load_frag(kp0),      s0);
    s0 = wmma_bf16(aq1, load_frag(kp0 + 32), s0);
    s1 = wmma_bf16(aq0, load_frag(kp1),      s1);
    s1 = wmma_bf16(aq1, load_frag(kp1 + 32), s1);

    // current tile's TDM must be complete (in-order TENSORcnt)
#if HAVE_TDM
    if (more) __builtin_amdgcn_s_wait_tensorcnt(1);
    else      __builtin_amdgcn_s_wait_tensorcnt(0);
#endif

    const int kc0 = kt + l15, kc1 = kc0 + 16;
    const float mk0 = mask[b * QLEN + kc0];
    const float mk1 = mask[b * QLEN + kc1];

    float p0[8], p1[8], alpha[8];
#pragma unroll
    for (int r = 0; r < 8; ++r) {
      const int qr = qRow0 + r + 8 * hi16;
      const int rr = r + 8 * hi16;
#if HAVE_TDM
      float bi0 = bias_t[wave][buf][rr * 32 + l15];
      float bi1 = bias_t[wave][buf][rr * 32 + 16 + l15];
#else
      float bi0 = __builtin_nontemporal_load(
          &bias_base[(size_t)rr * QLEN + kc0]);
      float bi1 = __builtin_nontemporal_load(
          &bias_base[(size_t)rr * QLEN + kc1]);
#endif
      float sv0 = s0[r] * 0.125f + bi0 + mk0 + (kc0 > qr ? NEGINF : 0.0f);
      float sv1 = s1[r] * 0.125f + bi1 + mk1 + (kc1 > qr ? NEGINF : 0.0f);

      // all-lane row reductions on DPP16 rotates (pure VALU)
      float rmax = row_allmax(fmaxf(sv0, sv1));
      float m_new = fmaxf(m_run[r], rmax);
      float a = __expf(m_run[r] - m_new);
      p0[r] = __expf(sv0 - m_new);
      p1[r] = __expf(sv1 - m_new);
      float rsum = row_allsum(p0[r] + p1[r]);
      l_run[r] = l_run[r] * a + rsum;
      m_run[r] = m_new;
      alpha[r] = a;
    }
#pragma unroll
    for (int nt = 0; nt < 4; ++nt)
#pragma unroll
      for (int r = 0; r < 8; ++r) o_acc[nt][r] *= alpha[r];

    // ---- P: C-layout -> A-layout via per-wave LDS tile
#pragma unroll
    for (int r = 0; r < 8; ++r) {
      lds_p[wave][r + 8 * hi16][l15]      = (bf16_t)p0[r];
      lds_p[wave][r + 8 * hi16][16 + l15] = (bf16_t)p1[r];
    }
    asm volatile("s_wait_dscnt 0" ::: "memory");
    v16bf pa;
    {
      v8bf lo = *(const v8bf*)&lds_p[wave][l15][kb];
      v8bf hi = *(const v8bf*)&lds_p[wave][l15][kb + 16];
#pragma unroll
      for (int i = 0; i < 8; ++i) { pa[i] = lo[i]; pa[i + 8] = hi[i]; }
    }

    // ---- O += P V  (V pre-transposed: contiguous B-fragments)
    const bf16_t* vp =
        vt + ((size_t)(bh * DK) << 11) + kt + kb;  // + d*2048 per column
#pragma unroll
    for (int nt = 0; nt < 4; ++nt) {
      v16bf bv = load_frag(vp + ((size_t)(nt * 16 + l15) << 11));
      o_acc[nt] = wmma_bf16(pa, bv, o_acc[nt]);
    }
    buf ^= 1;
  }

  // ---- epilogue: normalize and store bf16
  float inv_l[8];
#pragma unroll
  for (int r = 0; r < 8; ++r) inv_l[r] = 1.0f / l_run[r];
#pragma unroll
  for (int nt = 0; nt < 4; ++nt)
#pragma unroll
    for (int r = 0; r < 8; ++r) {
      int qr  = qRow0 + r + 8 * hi16;
      int col = h * DK + nt * 16 + l15;
      obf[((size_t)(b * QLEN) + qr) * HD + col] =
          (bf16_t)(o_acc[nt][r] * inv_l[r]);
    }
}

// ---------------------------------------------------------------- launcher
extern "C" void kernel_launch(void* const* d_in, const int* in_sizes, int n_in,
                              void* d_out, int out_size, void* d_ws,
                              size_t ws_size, hipStream_t stream) {
  const float* hidden = (const float*)d_in[0];
  const float* amask  = (const float*)d_in[1];
  const float* pbias  = (const float*)d_in[2];
  const float* Wq     = (const float*)d_in[3];
  const float* Wk     = (const float*)d_in[4];
  const float* Wv     = (const float*)d_in[5];
  const float* Wo     = (const float*)d_in[6];

  char* ws = (char*)d_ws;
  size_t off = 0;
  auto alloc = [&](size_t elems) {
    bf16_t* p = (bf16_t*)(ws + off);
    off += ((elems * sizeof(bf16_t)) + 255) & ~(size_t)255;
    return p;
  };
  const size_t actElems = (size_t)MTOT * HD;   // 3,145,728
  const size_t wElems   = (size_t)HD * HD;     //   589,824
  bf16_t* hx   = alloc(actElems);
  bf16_t* q_bf = alloc(actElems);
  bf16_t* k_bf = alloc(actElems);
  bf16_t* v_t  = alloc(actElems);
  bf16_t* attn = alloc(actElems);
  bf16_t* wqt  = alloc(wElems);
  bf16_t* wkt  = alloc(wElems);
  bf16_t* wvt  = alloc(wElems);
  bf16_t* wot  = alloc(wElems);

  // 1) convert hidden, transpose+convert weights
  cvt_f32_to_bf16<<<(int)((actElems + 255) / 256), 256, 0, stream>>>(
      hidden, hx, (int)actElems);
  transpose_w_bf16<<<(int)(wElems / 256), 256, 0, stream>>>(Wq, wqt);
  transpose_w_bf16<<<(int)(wElems / 256), 256, 0, stream>>>(Wk, wkt);
  transpose_w_bf16<<<(int)(wElems / 256), 256, 0, stream>>>(Wv, wvt);
  transpose_w_bf16<<<(int)(wElems / 256), 256, 0, stream>>>(Wo, wot);

  // 2) QKV projections
  dim3 ggrid(HD / 128, MTOT / 64);
  gemm_bf16<0><<<ggrid, 128, 0, stream>>>(hx, wqt, q_bf);
  gemm_bf16<0><<<ggrid, 128, 0, stream>>>(hx, wkt, k_bf);
  gemm_bf16<1><<<ggrid, 128, 0, stream>>>(hx, wvt, v_t);

  // 3) fused attention (flash-style, causal, bias+mask, TDM bias prefetch)
  attn_kernel<<<dim3(QLEN / 64, BATCH * NH), 128, 0, stream>>>(
      q_bf, k_bf, v_t, pbias, amask, attn);

  // 4) output projection -> f32
  gemm_bf16<2><<<ggrid, 128, 0, stream>>>(attn, wot, d_out);
}
